// MultiHeadAttention_48369921687755
// MI455X (gfx1250) — matrix operand order repaired and checked by reading, not compiled, not run on hardware
//
#include <hip/hip_runtime.h>

#define B_  2
#define S_  2048
#define H_  1024
#define NH_ 16
#define D_  64

typedef unsigned int u32;
typedef __attribute__((ext_vector_type(16))) __bf16 v16bf;
typedef __attribute__((ext_vector_type(8)))  __bf16 v8bf;
typedef __attribute__((ext_vector_type(8)))  float  v8f;
typedef __attribute__((ext_vector_type(4)))  u32    u32x4;
typedef __attribute__((ext_vector_type(8)))  int    i32x8;
typedef __attribute__((ext_vector_type(4)))  int    i32x4;

union FragU { v16bf v; v8bf h[2]; };

__device__ __forceinline__ unsigned short f2bf(float f) {
  union { float f; unsigned u; } x; x.f = f;
  unsigned r = x.u + 0x7FFFu + ((x.u >> 16) & 1u);   // round-to-nearest-even
  return (unsigned short)(r >> 16);
}

// ---------------------------------------------------------------------------
// TDM: 2-D tile load (global -> LDS). data_size = 3 (8-byte units).
// Issue from ONE wave only (tensor ops ignore EXEC), defer s_wait_tensorcnt
// to overlap the DMA with compute, then workgroup-barrier before consumers.
// ---------------------------------------------------------------------------
__device__ __forceinline__ void tdm_load_2d(u32 lds_off, const void* gptr,
                                            u32 row_bytes, u32 rows,
                                            u32 row_pitch_bytes) {
  const u32 units  = row_bytes >> 3;        // tile_dim0 / tensor_dim0 (8B units)
  const u32 stride = row_pitch_bytes >> 3;  // tensor_dim0_stride (8B units)
  const unsigned long long ga = (unsigned long long)(uintptr_t)gptr;

  u32x4 g0;
  g0[0] = 1u;                                            // count=1, user mode
  g0[1] = lds_off;                                       // LDS byte address
  g0[2] = (u32)ga;                                       // global_addr[31:0]
  g0[3] = (u32)((ga >> 32) & 0x01FFFFFFu) | (2u << 30);  // addr[56:32] | type=2

  i32x8 g1;
  g1[0] = (int)(3u << 16);                               // data_size=3, mask=0
  g1[1] = (int)((units & 0xFFFFu) << 16);                // tensor_dim0 lo
  g1[2] = (int)(((units >> 16) & 0xFFFFu) | ((rows & 0xFFFFu) << 16));
  g1[3] = (int)(((rows >> 16) & 0xFFFFu) | ((units & 0xFFFFu) << 16));
  g1[4] = (int)(rows & 0xFFFFu);                         // tile_dim1
  g1[5] = (int)stride;                                   // dim0_stride[31:0]
  g1[6] = 0;
  g1[7] = 0;

  i32x4 g2 = {0, 0, 0, 0};
  i32x4 g3 = {0, 0, 0, 0};
#if defined(__clang_major__) && (__clang_major__ >= 23)
  i32x8 g4 = {0, 0, 0, 0, 0, 0, 0, 0};
  __builtin_amdgcn_tensor_load_to_lds(g0, g1, g2, g3, g4, 0);
#else
  __builtin_amdgcn_tensor_load_to_lds(g0, g1, g2, g3, 0);
#endif
}

// A-fragment (16 rows x 32 K): lane L holds row M=L%16,
// K = 8*(L>=16) + e + 8*(e>=8)  ->  two 16B chunks from a row-major tile.
__device__ __forceinline__ v16bf load_fragA(const unsigned short* lds, int stride,
                                            int rowBase, int kBase, int lane) {
  int row  = rowBase + (lane & 15);
  int base = kBase + ((lane >> 4) & 1) * 8;
  const unsigned short* p = lds + row * stride + base;
  FragU f;
  f.h[0] = *(const v8bf*)(p);
  f.h[1] = *(const v8bf*)(p + 16);
  return f.v;
}
// B-fragment (32 K x 16 cols): lane L holds col N=L%16,
// K = 8*(L>=16) + e + 8*(e>=8)  ->  two 16B chunks from an N-major tile.
__device__ __forceinline__ v16bf load_fragB(const unsigned short* lds, int stride,
                                            int rowBase, int kBase, int lane) {
  int row = rowBase + (lane & 15);
  int off = kBase + ((lane >> 4) & 1) * 8;
  const unsigned short* p = lds + row * stride + off;
  FragU f;
  f.h[0] = *(const v8bf*)(p);
  f.h[1] = *(const v8bf*)(p + 16);
  return f.v;
}

__device__ __forceinline__ v8f wmma_bf16(v16bf a, v16bf b, v8f c) {
  return __builtin_amdgcn_wmma_f32_16x16x32_bf16(false, a, false, b, (short)0, c,
                                                 false, false);
}

// ---------------------------------------------------------------------------
// Prep 1: x (fp32, 4096x1024) -> bf16, row-major. One-time streaming pass.
// ---------------------------------------------------------------------------
__global__ __launch_bounds__(256) void convert_x_kernel(
    const float* __restrict__ x, unsigned short* __restrict__ xb) {
  size_t i = ((size_t)blockIdx.x * 256 + threadIdx.x) * 4;
  float4 v = *(const float4*)(x + i);
  uint2 pk;
  pk.x = (u32)f2bf(v.x) | ((u32)f2bf(v.y) << 16);
  pk.y = (u32)f2bf(v.z) | ((u32)f2bf(v.w) << 16);
  *(uint2*)(xb + i) = pk;
}

// ---------------------------------------------------------------------------
// Prep 2: W (fp32, 1024x1024, K-major) -> bf16 transposed (N-major), so GEMM
// B-tiles are contiguous-K rows and can be DMA'd by the TDM.
// ---------------------------------------------------------------------------
__global__ __launch_bounds__(256) void transpose_w_kernel(
    const float* __restrict__ W0, const float* __restrict__ W1,
    const float* __restrict__ W2, const float* __restrict__ W3,
    unsigned short* __restrict__ Tbase) {
  const float* W = W0;
  if (blockIdx.z == 1) W = W1;
  else if (blockIdx.z == 2) W = W2;
  else if (blockIdx.z == 3) W = W3;
  unsigned short* WT = Tbase + (size_t)blockIdx.z * H_ * H_;

  int i = blockIdx.x * 256 + threadIdx.x;   // 0..262143
  int k = i >> 8;                           // 0..1023
  int n4 = (i & 255) << 2;                  // 0..1020
  float4 v = *(const float4*)(W + (size_t)k * H_ + n4);
  WT[(size_t)(n4 + 0) * H_ + k] = f2bf(v.x);
  WT[(size_t)(n4 + 1) * H_ + k] = f2bf(v.y);
  WT[(size_t)(n4 + 2) * H_ + k] = f2bf(v.z);
  WT[(size_t)(n4 + 3) * H_ + k] = f2bf(v.w);
}

// ---------------------------------------------------------------------------
// GEMM: C(4096x1024) = A_bf16 @ W  (W pre-transposed bf16, N-major).
// Tile 64(M) x 128(N), 128 threads = 4 waves; wave owns strips wave*16 and
// 64+wave*16 -> 8 WMMAs per K-step. Both tiles staged by TDM with LDS
// double-buffering: issue next slab's DMAs, compute, THEN wait tensorcnt.
// MODE 0: outputs Q/K head-split bf16 and V transposed [b,h][d][s] bf16.
// MODE 1: output fp32 row-major (final projection).
// ---------------------------------------------------------------------------
template <int MODE>
__global__ __launch_bounds__(128) void gemm_bf16_kernel(
    const unsigned short* __restrict__ A,
    const unsigned short* __restrict__ T0, const unsigned short* __restrict__ T1,
    const unsigned short* __restrict__ T2,
    unsigned short* __restrict__ Oq, unsigned short* __restrict__ Ok,
    unsigned short* __restrict__ Ovt, float* __restrict__ Of) {
  __shared__ unsigned short sA[2][64 * 32];    // row-major M x K
  __shared__ unsigned short sB[2][128 * 32];   // N-major  N x K

  const int tid  = threadIdx.x;
  const int wave = tid >> 5;
  const int lane = tid & 31;
  const int n0 = blockIdx.x * 128;
  const int m0 = blockIdx.y * 64;
  const int which = (MODE == 0) ? (int)blockIdx.z : 0;
  const unsigned short* WT = T0;
  if (MODE == 0) {
    if (which == 1) WT = T1;
    else if (which == 2) WT = T2;
  }

  // prologue: DMA K-slab 0 into buffer 0
  if (wave == 0) {
    tdm_load_2d((u32)(uintptr_t)(void*)&sA[0][0], A + (size_t)m0 * H_,
                /*row_bytes=*/64, /*rows=*/64, /*pitch=*/H_ * 2);
    tdm_load_2d((u32)(uintptr_t)(void*)&sB[0][0], WT + (size_t)n0 * H_,
                64, 128, H_ * 2);
    __builtin_amdgcn_s_wait_tensorcnt(0);
  }
  __syncthreads();

  v8f acc[2][4] = {{{}, {}, {}, {}}, {{}, {}, {}, {}}};

  for (int it = 0; it < H_ / 32; ++it) {
    const int cur = it & 1;
    if (it + 1 < H_ / 32 && wave == 0) {     // async: next slab into other buffer
      const int kb = (it + 1) * 32;
      tdm_load_2d((u32)(uintptr_t)(void*)&sA[cur ^ 1][0],
                  A + (size_t)m0 * H_ + kb, 64, 64, H_ * 2);
      tdm_load_2d((u32)(uintptr_t)(void*)&sB[cur ^ 1][0],
                  WT + (size_t)n0 * H_ + kb, 64, 128, H_ * 2);
    }

    v16bf bf0 = load_fragB(sB[cur], 32, wave * 16, 0, lane);
    v16bf bf1 = load_fragB(sB[cur], 32, 64 + wave * 16, 0, lane);
#pragma unroll
    for (int ms = 0; ms < 4; ++ms) {
      v16bf af = load_fragA(sA[cur], 32, ms * 16, 0, lane);
      acc[0][ms] = wmma_bf16(af, bf0, acc[0][ms]);
      acc[1][ms] = wmma_bf16(af, bf1, acc[1][ms]);
    }

    if (it + 1 < H_ / 32 && wave == 0) __builtin_amdgcn_s_wait_tensorcnt(0);
    __syncthreads();
  }

  // ---- epilogue: C layout: VGPR r, lane L -> M = r + 8*(L/16), N = L%16 ----
#pragma unroll
  for (int s2 = 0; s2 < 2; ++s2) {
    const int n = n0 + s2 * 64 + wave * 16 + (lane & 15);
    const int h = n >> 6, d = n & 63;
#pragma unroll
    for (int ms = 0; ms < 4; ++ms) {
      const int mb = m0 + ms * 16 + ((lane >> 4) << 3);   // r = 0 row
      if (MODE == 1) {
#pragma unroll
        for (int r = 0; r < 8; ++r)
          Of[(size_t)(mb + r) * H_ + n] = acc[s2][ms][r];
      } else if (which == 2) {
        // V transposed: [b,h][d][s]; consecutive r -> consecutive s: pack b32
        const int bb = mb >> 11, sb = mb & (S_ - 1);
        unsigned short* dst = Ovt + ((size_t)(bb * NH_ + h) * D_ + d) * S_ + sb;
#pragma unroll
        for (int r = 0; r < 8; r += 2) {
          u32 pk = (u32)f2bf(acc[s2][ms][r]) |
                   ((u32)f2bf(acc[s2][ms][r + 1]) << 16);
          *(u32*)(dst + r) = pk;
        }
      } else {
        unsigned short* O = (which == 1) ? Ok : Oq;
        const int bb = mb >> 11, sb = mb & (S_ - 1);
#pragma unroll
        for (int r = 0; r < 8; ++r)
          O[((size_t)(bb * NH_ + h) * S_ + sb + r) * D_ + d] = f2bf(acc[s2][ms][r]);
      }
    }
  }
}

// ---------------------------------------------------------------------------
// Flash attention: block = (64-query tile, h, b), 128 threads = 4 waves,
// wave w owns query rows [w*16, w*16+16). Causal, online softmax.
// Q, K and V^T tiles are ALL staged by TDM; K/V^T double-buffered with the
// tensor-wait deferred past the WMMA work. 1/sqrt(d) folded into exp2 scale.
// ---------------------------------------------------------------------------
__global__ __launch_bounds__(128) void flash_attn_kernel(
    const unsigned short* __restrict__ Q, const unsigned short* __restrict__ Km,
    const unsigned short* __restrict__ VT, unsigned short* __restrict__ Oa) {
  __shared__ unsigned short sQ[64 * 64];         // q-rows x d
  __shared__ unsigned short sK[2][64 * 64];      // k-rows x d   (B-frag: K^T)
  __shared__ unsigned short sVT[2][64 * 64];     // d x k-rows   (B-frag: V)
  __shared__ unsigned short sP[4 * 16 * 64];     // per-wave P tile, row-major

  const int tid  = threadIdx.x;
  const int wave = tid >> 5;
  const int lane = tid & 31;
  const int qt = blockIdx.x;                     // 0..31
  const int h = blockIdx.y, b = blockIdx.z;
  const size_t bh  = (size_t)(b * NH_ + h) * S_ * D_;   // Q/K base
  const size_t vtb = (size_t)(b * NH_ + h) * D_ * S_;   // V^T base [d][s]
  const float SCL = 0.125f * 1.44269504089f;     // (1/sqrt(64)) * log2(e)

  // ---- prologue: DMA Q tile + key-tile 0 (K and V^T) ----
  if (wave == 0) {
    tdm_load_2d((u32)(uintptr_t)(void*)sQ, Q + bh + (size_t)qt * 64 * D_,
                /*row_bytes=*/128, /*rows=*/64, /*pitch=*/128);
    tdm_load_2d((u32)(uintptr_t)(void*)&sK[0][0], Km + bh, 128, 64, 128);
    tdm_load_2d((u32)(uintptr_t)(void*)&sVT[0][0], VT + vtb, 128, 64, S_ * 2);
    __builtin_amdgcn_s_wait_tensorcnt(0);
  }
  __syncthreads();

  v16bf qf[2];
#pragma unroll
  for (int kc = 0; kc < 2; ++kc)
    qf[kc] = load_fragA(sQ, 64, wave * 16, kc * 32, lane);

  float mrow[8], lrow[8];
#pragma unroll
  for (int r = 0; r < 8; ++r) { mrow[r] = -3.0e38f; lrow[r] = 0.f; }
  v8f acco[4] = {{}, {}, {}, {}};

  for (int kt = 0; kt <= qt; ++kt) {
    const int cur = kt & 1;
    if (kt < qt && wave == 0) {     // async: next key tile into other buffers
      tdm_load_2d((u32)(uintptr_t)(void*)&sK[cur ^ 1][0],
                  Km + bh + (size_t)(kt + 1) * 64 * D_, 128, 64, 128);
      tdm_load_2d((u32)(uintptr_t)(void*)&sVT[cur ^ 1][0],
                  VT + vtb + (size_t)(kt + 1) * 64, 128, 64, S_ * 2);
    }

    // ---- S = Q @ K^T  (16q x 64k per wave) ----
    v8f sacc[4] = {{}, {}, {}, {}};
#pragma unroll
    for (int kc = 0; kc < 2; ++kc) {
#pragma unroll
      for (int t = 0; t < 4; ++t) {
        v16bf kf = load_fragB(sK[cur], 64, t * 16, kc * 32, lane);
        sacc[t] = wmma_bf16(qf[kc], kf, sacc[t]);
      }
    }

    // ---- causal mask (diagonal tile only) ----
    if (kt == qt) {
#pragma unroll
      for (int t = 0; t < 4; ++t) {
#pragma unroll
        for (int r = 0; r < 8; ++r) {
          int kl = t * 16 + (lane & 15);
          int ql = wave * 16 + r + ((lane >> 4) << 3);
          if (kl > ql) sacc[t][r] = -3.0e38f;
        }
      }
    }

    // ---- online softmax (row stats across each 16-lane half) ----
#pragma unroll
    for (int r = 0; r < 8; ++r) {
      float mx = fmaxf(fmaxf(sacc[0][r], sacc[1][r]),
                       fmaxf(sacc[2][r], sacc[3][r]));
#pragma unroll
      for (int off = 1; off < 16; off <<= 1) mx = fmaxf(mx, __shfl_xor(mx, off, 16));
      float mn = fmaxf(mrow[r], mx);
      float alpha = exp2f((mrow[r] - mn) * SCL);
      float sum = 0.f;
#pragma unroll
      for (int t = 0; t < 4; ++t) {
        float p = exp2f((sacc[t][r] - mn) * SCL);
        sacc[t][r] = p;
        sum += p;
      }
#pragma unroll
      for (int off = 1; off < 16; off <<= 1) sum += __shfl_xor(sum, off, 16);
      lrow[r] = lrow[r] * alpha + sum;
      mrow[r] = mn;
#pragma unroll
      for (int t = 0; t < 4; ++t) acco[t][r] *= alpha;
      // C-layout -> row-major bf16 P in this wave's private LDS region
      int prow = r + ((lane >> 4) << 3);
#pragma unroll
      for (int t = 0; t < 4; ++t)
        sP[wave * 1024 + prow * 64 + t * 16 + (lane & 15)] = f2bf(sacc[t][r]);
    }

    // ---- O += P @ V  (P read back as A-fragments; same-wave DS ordering) ----
#pragma unroll
    for (int kc = 0; kc < 2; ++kc) {
      v16bf pf = load_fragA(&sP[wave * 1024], 64, 0, kc * 32, lane);
#pragma unroll
      for (int t = 0; t < 4; ++t) {
        v16bf vf = load_fragB(sVT[cur], 64, t * 16, kc * 32, lane);
        acco[t] = wmma_bf16(pf, vf, acco[t]);
      }
    }

    if (kt < qt && wave == 0) __builtin_amdgcn_s_wait_tensorcnt(0);
    __syncthreads();
  }

  // ---- epilogue: normalize, write merged-head bf16 (B,S,H) ----
#pragma unroll
  for (int t = 0; t < 4; ++t) {
#pragma unroll
    for (int r = 0; r < 8; ++r) {
      int ql = wave * 16 + r + ((lane >> 4) << 3);
      int q = qt * 64 + ql;
      float o = acco[t][r] / lrow[r];
      size_t idx = (size_t)(b * S_ + q) * H_ + h * D_ + t * 16 + (lane & 15);
      Oa[idx] = f2bf(o);
    }
  }
}

// ---------------------------------------------------------------------------
extern "C" void kernel_launch(void* const* d_in, const int* in_sizes, int n_in,
                              void* d_out, int out_size, void* d_ws, size_t ws_size,
                              hipStream_t stream) {
  (void)in_sizes; (void)n_in; (void)out_size; (void)ws_size;
  const float* x  = (const float*)d_in[0];
  // d_in[1]: causal mask (bool) — causality is hardcoded in flash_attn_kernel
  const float* Wq = (const float*)d_in[2];
  const float* Wk = (const float*)d_in[3];
  const float* Wv = (const float*)d_in[4];
  const float* Wo = (const float*)d_in[5];

  const size_t MS = (size_t)4096 * 1024;       // 4M elements
  const size_t WS1 = (size_t)H_ * H_;          // 1M elements per weight
  unsigned short* ws = (unsigned short*)d_ws;
  unsigned short* xb    = ws;                  // bf16 x           (4M)
  unsigned short* wt    = ws + MS;             // 4x bf16 W^T      (4M)
  unsigned short* q_ws  = ws + 2 * MS;         // head-split Q     (4M)
  unsigned short* k_ws  = ws + 3 * MS;         // head-split K     (4M)
  unsigned short* vt_ws = ws + 4 * MS;         // [b,h][d][s] V^T  (4M)
  unsigned short* a_ws  = ws + 5 * MS;         // merged attn out  (4M)

  // 0) one-time format conversions (streaming, ~2us at HBM rate)
  convert_x_kernel<<<dim3((B_ * S_ * H_) / 1024), 256, 0, stream>>>(x, xb);
  transpose_w_kernel<<<dim3((H_ * H_) / 1024, 1, 4), 256, 0, stream>>>(
      Wq, Wk, Wv, Wo, wt);

  // 1) Q/K/V projections (TDM-fed, double-buffered, bf16 WMMA)
  gemm_bf16_kernel<0><<<dim3(H_ / 128, (B_ * S_) / 64, 3), 128, 0, stream>>>(
      xb, wt, wt + WS1, wt + 2 * WS1, q_ws, k_ws, vt_ws, nullptr);
  // 2) fused causal flash attention (Q/K/V^T all TDM-staged, double-buffered)
  flash_attn_kernel<<<dim3(S_ / 64, NH_, B_), 128, 0, stream>>>(q_ws, k_ws, vt_ws,
                                                                a_ws);
  // 3) output projection -> fp32 d_out
  gemm_bf16_kernel<1><<<dim3(H_ / 128, (B_ * S_) / 64, 1), 128, 0, stream>>>(
      a_ws, wt + 3 * WS1, nullptr, nullptr, nullptr, nullptr, nullptr,
      (float*)d_out);
}
